// UpdateU_11089605559054
// MI455X (gfx1250) — matrix operand order repaired and checked by reading, not compiled, not run on hardware
//
#include <hip/hip_runtime.h>

// ---------------------------------------------------------------------------
// Fused 2-layer MLP + segment-mean for MI455X (gfx1250, wave32, WMMA).
//
//   h  = LeakyReLU(v @ W1 + b1, 0.015);  h2 = h @ W2 + b2
//   out[g] = mean over rows r with batch[r]==g of h2[r]   (batch sorted)
//
// Design (see analysis): memory floor ~44us (1GB v reads @ 23.3TB/s),
// 103 GFLOP in bf16 WMMA ~40us -> must fuse both GEMMs + reduction in one
// pass, keep weights in LDS, never spill h to HBM.
// ---------------------------------------------------------------------------

typedef __bf16 bf16_t;
typedef __attribute__((ext_vector_type(16))) __bf16       v16bf;
typedef __attribute__((ext_vector_type(8)))  float        v8f;
typedef __attribute__((ext_vector_type(4)))  float        v4f;
typedef __attribute__((ext_vector_type(4)))  unsigned int v4u;

#define HIN   256   // input features  (K of layer 1)
#define HMID  128   // hidden features (N of layer 1, K of layer 2)
#define OCH   128   // output features (N of layer 2)
#define ROWS  128   // v rows per block (8 waves x 16 rows)
#define NEG_SLOPE 0.015f

// LDS pitches (elements), padded to spread banks; all row strides are
// multiples of 16B so v4u (b128) LDS accesses stay aligned.
#define W1P   264   // bf16: 256 + 8   (row = 528B)
#define W2P   136   // bf16: 128 + 8   (row = 272B)
#define SHP   136   // bf16 h1 staging
#define SOP   132   // f32 h2 tile: 128 + 4 (row = 528B, overlays sW1t exactly)

#define SW1_BYTES  (HMID * W1P * 2)            // 67584, also == ROWS*SOP*4
#define SW2_BYTES  (HMID * W2P * 2)            // 34816
#define SH_BYTES   (ROWS * SHP * 2)            // 34816
#define SSEG_BYTES (ROWS * 4)                  // 512
#define SMEM_BYTES (SW1_BYTES + SW2_BYTES + SH_BYTES + SSEG_BYTES) // 137728 < 320KB/WGP

union ABf { v16bf v; v4u u[2]; __bf16 e[16]; };

__global__ void __launch_bounds__(256, 2)
fused_mlp_segmean(const float* __restrict__ v,
                  const int*   __restrict__ batch,
                  const float* __restrict__ W1,   // [HIN][HMID] row-major
                  const float* __restrict__ b1,   // [HMID]
                  const float* __restrict__ W2,   // [HMID][OCH] row-major
                  const float* __restrict__ b2,   // [OCH]
                  float* __restrict__ accum,      // [G][OCH] zero-initialized
                  float* __restrict__ cnt)        // [G]      zero-initialized
{
    extern __shared__ char smem[];
    bf16_t* sW1t = (bf16_t*)(smem);                        // [n][k] transposed W1 (bf16)
    float*  sOut = (float*) (smem);                        // overlays sW1t after layer 1
    bf16_t* sW2t = (bf16_t*)(smem + SW1_BYTES);            // [n][k] transposed W2 (bf16)
    bf16_t* sH   = (bf16_t*)(smem + SW1_BYTES + SW2_BYTES);// [row][k] h1 staging (bf16)
    int*    sSeg = (int*)   (smem + SW1_BYTES + SW2_BYTES + SH_BYTES);

    const int tid  = threadIdx.x;
    const int lane = tid & 31;
    const int wave = tid >> 5;
    const int hf   = lane >> 4;      // half-wave: selects A/B K-split per ISA layout
    const int l16  = lane & 15;
    const int rowBase = blockIdx.x * ROWS;
    const int myRow   = wave * 16 + l16;       // row this lane supplies to A-fragments

    // -------- stage weights (transposed, f32->bf16) and segment ids in LDS --
    for (int i = tid; i < HIN * HMID; i += 256) {
        int k = i >> 7, n = i & 127;                 // W1[k][n]
        sW1t[n * W1P + k] = (bf16_t)W1[i];
    }
    for (int i = tid; i < HMID * OCH; i += 256) {
        int k = i >> 7, n = i & 127;                 // W2[k][n]
        sW2t[n * W2P + k] = (bf16_t)W2[i];
    }
    if (tid < ROWS) sSeg[tid] = batch[rowBase + tid];
    __syncthreads();

    // ---------------- layer 1: h1 = LeakyReLU(v @ W1 + b1) ------------------
    // A (16x32 bf16, ISA 7.12.2): lane m=l16, e0..7 -> K=k0..k0+7,
    // e8..15 -> K=k0+16..k0+23 with k0 = kt*32 + hf*8.
    const float* vrow = v + (size_t)(rowBase + myRow) * HIN;

    v8f acc1[8];
#pragma unroll
    for (int nt = 0; nt < 8; ++nt) acc1[nt] = (v8f){};

#pragma unroll
    for (int kt = 0; kt < 8; ++kt) {
        const int k0 = kt * 32 + hf * 8;
        v4f f0 = *(const v4f*)(vrow + k0);
        v4f f1 = *(const v4f*)(vrow + k0 + 4);
        v4f f2 = *(const v4f*)(vrow + k0 + 16);
        v4f f3 = *(const v4f*)(vrow + k0 + 20);
        ABf a;
#pragma unroll
        for (int j = 0; j < 4; ++j) {
            a.e[j]      = (__bf16)f0[j];
            a.e[4 + j]  = (__bf16)f1[j];
            a.e[8 + j]  = (__bf16)f2[j];
            a.e[12 + j] = (__bf16)f3[j];
        }
#pragma unroll
        for (int nt = 0; nt < 8; ++nt) {
            // B (32x16 bf16): lane n = nt*16+l16, same K order as A (k0..k0+7, k0+16..k0+23)
            const bf16_t* bp = sW1t + (nt * 16 + l16) * W1P + kt * 32 + hf * 8;
            ABf b;
            b.u[0] = *(const v4u*)(bp);
            b.u[1] = *(const v4u*)(bp + 16);
            acc1[nt] = __builtin_amdgcn_wmma_f32_16x16x32_bf16(
                false, a.v, false, b.v, (short)0, acc1[nt], false, false);
        }
    }

    // bias + LeakyReLU, scatter C-layout (M = r + 8*hf, N = l16) into sH rows (bf16)
#pragma unroll
    for (int nt = 0; nt < 8; ++nt) {
        const int   col = nt * 16 + l16;
        const float bb  = b1[col];
#pragma unroll
        for (int r = 0; r < 8; ++r) {
            float x = acc1[nt][r] + bb;
            x = (x >= 0.0f) ? x : NEG_SLOPE * x;
            sH[(wave * 16 + r + 8 * hf) * SHP + col] = (bf16_t)x;
        }
    }
    __syncthreads();   // all waves done with sW1t + sH fully written

    // ---------------- layer 2: h2 = h1 @ W2 + b2 ----------------------------
    v8f acc2[8];
#pragma unroll
    for (int nt = 0; nt < 8; ++nt) acc2[nt] = (v8f){};

    const bf16_t* hrow = sH + myRow * SHP;
#pragma unroll
    for (int kt = 0; kt < 4; ++kt) {
        const int k0 = kt * 32 + hf * 8;
        ABf a;
        a.u[0] = *(const v4u*)(hrow + k0);
        a.u[1] = *(const v4u*)(hrow + k0 + 16);
#pragma unroll
        for (int nt = 0; nt < 8; ++nt) {
            const bf16_t* bp = sW2t + (nt * 16 + l16) * W2P + kt * 32 + hf * 8;
            ABf b;
            b.u[0] = *(const v4u*)(bp);
            b.u[1] = *(const v4u*)(bp + 16);
            acc2[nt] = __builtin_amdgcn_wmma_f32_16x16x32_bf16(
                false, a.v, false, b.v, (short)0, acc2[nt], false, false);
        }
    }

    // write f32 h2 tile to LDS (overlays sW1t region; safe: everyone is past layer 1)
#pragma unroll
    for (int nt = 0; nt < 8; ++nt) {
        const int   col = nt * 16 + l16;
        const float bb  = b2[col];
#pragma unroll
        for (int r = 0; r < 8; ++r)
            sOut[(wave * 16 + r + 8 * hf) * SOP + col] = acc2[nt][r] + bb;
    }
    __syncthreads();

    // ---------------- segment reduction (batch sorted => runs) --------------
    if (tid < ROWS) atomicAdd(&cnt[sSeg[tid]], 1.0f);

    {   // each thread owns one column over half the rows; flush one atomic per run
        const int col = tid & 127;
        const int r0  = (tid >> 7) * 64;
        int   cur = sSeg[r0];
        float s   = 0.0f;
        for (int r = r0; r < r0 + 64; ++r) {
            const int sg = sSeg[r];
            if (sg != cur) {
                atomicAdd(&accum[(size_t)cur * OCH + col], s);
                s = 0.0f; cur = sg;
            }
            s += sOut[r * SOP + col];
        }
        atomicAdd(&accum[(size_t)cur * OCH + col], s);
    }
}

__global__ void zero_ws(float* __restrict__ p, int n) {
    int i = blockIdx.x * blockDim.x + threadIdx.x;
    if (i < n) p[i] = 0.0f;
}

__global__ void finalize_mean(const float* __restrict__ accum,
                              const float* __restrict__ cnt,
                              float* __restrict__ out, int total) {
    int i = blockIdx.x * blockDim.x + threadIdx.x;
    if (i < total) out[i] = accum[i] / fmaxf(cnt[i >> 7], 1.0f);
}

extern "C" void kernel_launch(void* const* d_in, const int* in_sizes, int n_in,
                              void* d_out, int out_size, void* d_ws, size_t ws_size,
                              hipStream_t stream) {
    const float* v     = (const float*)d_in[0];
    const int*   batch = (const int*)  d_in[1];  // jnp.int64 -> int32 (x64 disabled)
    // d_in[2] = num_graphs scalar on device (unused; G derived from out_size)
    const float* W1 = (const float*)d_in[3];
    const float* b1 = (const float*)d_in[4];
    const float* W2 = (const float*)d_in[5];
    const float* b2 = (const float*)d_in[6];
    float* out = (float*)d_out;

    const int nrows = in_sizes[0] / HIN;   // N = 1048576
    const int G     = out_size / OCH;      // 8192

    float* accum = (float*)d_ws;           // G*OCH sums
    float* cnt   = accum + (size_t)G * OCH;// G counts  (total ~4.03MB of ws)

    // allow >64KB dynamic LDS (no-op if unnecessary; not a stream op, capture-safe)
    (void)hipFuncSetAttribute((const void*)fused_mlp_segmean,
                              hipFuncAttributeMaxDynamicSharedMemorySize, SMEM_BYTES);

    const int zn = G * OCH + G;
    zero_ws<<<(zn + 255) / 256, 256, 0, stream>>>(accum, zn);

    const int nBlocks = nrows / ROWS;      // 8192 blocks x 256 threads (8 waves)
    fused_mlp_segmean<<<nBlocks, 256, SMEM_BYTES, stream>>>(
        v, batch, W1, b1, W2, b2, accum, cnt);

    finalize_mean<<<(out_size + 255) / 256, 256, 0, stream>>>(accum, cnt, out, out_size);
}
